// AttnBlock_46746424050164
// MI455X (gfx1250) — matrix operand order repaired and checked by reading, not compiled, not run on hardware
//
#include <hip/hip_runtime.h>

#define Bn 8
#define Cn 512
#define Ln 2048
#define Gn 4
#define LDQ 520
#define LDP 72

typedef __bf16 bfrag __attribute__((ext_vector_type(16)));
typedef __bf16 bh8   __attribute__((ext_vector_type(8)));
typedef float  v8f   __attribute__((ext_vector_type(8)));

__device__ __forceinline__ v8f wmma_bf16(bfrag a, bfrag b, v8f c) {
  // D = A(16x32) * B(32x16) + C, fp32 accum
  return __builtin_amdgcn_wmma_f32_16x16x32_bf16(false, a, false, b,
                                                 (short)0, c, false, false);
}

// A-operand fragment (16x32 bf16) from row-major [rows x K], row stride ld elems.
// Lane half 0: K 0-7 and 16-23 of row (lane&15); half 1: K 8-15 and 24-31.
__device__ __forceinline__ bfrag load_a_frag(const __bf16* p, int ld) {
  int lane = threadIdx.x & 31;
  const __bf16* q = p + (size_t)(lane & 15) * ld + ((lane >> 4) << 3);
  bh8 lo = *(const bh8*)q;
  bh8 hi = *(const bh8*)(q + 16);
  return __builtin_shufflevector(lo, hi, 0,1,2,3,4,5,6,7,8,9,10,11,12,13,14,15);
}

// B-operand fragment (32x16 bf16) loaded from the *transposed* array Bt [N x K]
// row-major: lane (lane&15) holds column n; half 0: K 0-7 and 16-23, half 1: K 8-15 and 24-31.
__device__ __forceinline__ bfrag load_bt_frag(const __bf16* p, int ld) {
  int lane = threadIdx.x & 31;
  const __bf16* q = p + (size_t)(lane & 15) * ld + ((lane >> 4) << 3);
  bh8 lo = *(const bh8*)q;
  bh8 hi = *(const bh8*)(q + 16);
  return __builtin_shufflevector(lo, hi, 0,1,2,3,4,5,6,7,8,9,10,11,12,13,14,15);
}

// Async DMA: 16 bytes global -> LDS, tracked on ASYNCcnt (CDNA5 VGLOBAL async).
// VDST operand carries the LDS byte address (low 32 bits of the generic
// pointer == LDS offset; shared aperture lives in the high word).
__device__ __forceinline__ void async_copy16(__bf16* lds, const __bf16* g) {
#if defined(__gfx1250__)
  unsigned int l32 = (unsigned int)(unsigned long long)lds;
  unsigned long long ga = (unsigned long long)g;
  asm volatile("global_load_async_to_lds_b128 %0, %1, off"
               :: "v"(l32), "v"(ga) : "memory");
#else
  *(bh8*)lds = *(const bh8*)g;
#endif
}
__device__ __forceinline__ void async_wait0() {
#if defined(__gfx1250__)
  asm volatile("s_wait_asynccnt 0x0" ::: "memory");
#endif
}

// ---------------- weight f32 -> bf16 ----------------
__global__ void __launch_bounds__(256) cvt_kernel(const float* __restrict__ s,
                                                  __bf16* __restrict__ d, int n) {
  int i = blockIdx.x * 256 + threadIdx.x;
  if (i < n) d[i] = (__bf16)s[i];
}

// ---------------- GroupNorm -> xn^T [B,L,C] bf16 ----------------
__global__ void __launch_bounds__(256) gn_kernel(const float* __restrict__ x,
    const float* __restrict__ w, const float* __restrict__ bias,
    __bf16* __restrict__ xnT) {
  __shared__ float rs[256], rss[256];
  const int bg = blockIdx.x;
  const int b = bg >> 2, g = bg & 3;
  const int tid = threadIdx.x;
  const int CG = Cn / Gn;              // 128
  const int N = CG * Ln;               // 262144
  const float* xp = x + ((size_t)(b * Cn + g * CG)) * Ln;
  float s = 0.f, ss = 0.f;
  for (int i = tid; i < N; i += 256) {
    float v = xp[i];
    s += v; ss += v * v;
  }
  rs[tid] = s; rss[tid] = ss;
  __syncthreads();
  for (int st = 128; st > 0; st >>= 1) {
    if (tid < st) { rs[tid] += rs[tid + st]; rss[tid] += rss[tid + st]; }
    __syncthreads();
  }
  const float mean = rs[0] / (float)N;
  const float var  = rss[0] / (float)N - mean * mean;
  const float rstd = rsqrtf(var + 1e-6f);
  for (int i = tid; i < N; i += 256) {
    int ch = i >> 11;                  // / 2048
    int l  = i & 2047;
    int c  = g * CG + ch;
    float v = (xp[i] - mean) * rstd * w[c] + bias[c];
    xnT[((size_t)(b * Ln + l)) * Cn + c] = (__bf16)v;
  }
}

// ---------------- generic GEMM: D[l,o] = A[l,:] . Wt[o,:] + bias ----------------
// mode 0: Q  -> bf16 [B,L,C], scaled by C^-0.5
// mode 1: K  -> bf16 [B,L,C]
// mode 2: V  -> bf16 [B,C,L] (transposed store, PV B-operand layout)
// mode 3: proj -> f32 [B,C,L] + residual
__global__ void __launch_bounds__(256) gemm_kernel(
    const __bf16* __restrict__ A, const __bf16* __restrict__ Wt,
    const float* __restrict__ bias, __bf16* __restrict__ outb,
    float* __restrict__ outf, const float* __restrict__ resid,
    float scale, int mode) {
  const int b = blockIdx.z;
  const int m0 = blockIdx.x * 64, n0 = blockIdx.y * 64;
  const int w = threadIdx.x >> 5, lane = threadIdx.x & 31;
  const int mi = w & 3, nh = w >> 2;   // wave: rows m0+mi*16, cols n0+nh*32
  const int half = lane >> 4;
  const __bf16* ar = A + ((size_t)(b * Ln + m0 + mi * 16)) * Cn;
  const __bf16* b0 = Wt + (size_t)(n0 + nh * 32) * Cn;
  const __bf16* b1 = b0 + (size_t)16 * Cn;
  v8f acc0, acc1;
#pragma unroll
  for (int r = 0; r < 8; ++r) { acc0[r] = 0.f; acc1[r] = 0.f; }
#pragma unroll 4
  for (int kk = 0; kk < Cn; kk += 32) {
    bfrag a = load_a_frag(ar + kk, Cn);
    acc0 = wmma_bf16(a, load_bt_frag(b0 + kk, Cn), acc0);
    acc1 = wmma_bf16(a, load_bt_frag(b1 + kk, Cn), acc1);
  }
#pragma unroll
  for (int t = 0; t < 2; ++t) {
    v8f acc = t ? acc1 : acc0;
    int col = n0 + nh * 32 + t * 16 + (lane & 15);
    float bv = bias[col];
#pragma unroll
    for (int r = 0; r < 8; ++r) {
      int row = m0 + mi * 16 + r + half * 8;
      float v = (acc[r] + bv) * scale;
      if (mode == 2) {
        outb[((size_t)(b * Cn + col)) * Ln + row] = (__bf16)v;
      } else if (mode == 3) {
        size_t idx = ((size_t)(b * Cn + col)) * Ln + row;
        outf[idx] = v + resid[idx];
      } else {
        outb[((size_t)(b * Ln + row)) * Cn + col] = (__bf16)v;
      }
    }
  }
}

// ---------------- fused flash attention ----------------
// grid: (L/32, B); 256 threads = 8 waves. Wave w: row slab mi=w&1 (16 rows),
// S-tile col / O col slab ni=w>>1. O acc: 16 rows x 128 cols = 8 v8f frags.
__global__ void __launch_bounds__(256) flash_kernel(
    const __bf16* __restrict__ Qg, const __bf16* __restrict__ Kg,
    const __bf16* __restrict__ Vt, __bf16* __restrict__ Og) {
  __shared__ __bf16 Qs[32 * LDQ];
  __shared__ __bf16 Ps[32 * LDP];
  __shared__ float red_m[32][4];
  __shared__ float red_s[32][4];
  __shared__ float row_m[32];
  __shared__ float row_l[32];
  __shared__ float row_a[32];

  const int b = blockIdx.y;
  const int i0 = blockIdx.x * 32;
  const int tid = threadIdx.x;
  const int w = tid >> 5;
  const int lane = tid & 31;
  const int half = lane >> 4;
  const int mi = w & 1;
  const int ni = w >> 1;

  { // stage Q tile 32x512 into LDS with async DMA (ASYNCcnt-tracked)
    const __bf16* src = Qg + ((size_t)(b * Ln + i0)) * Cn;
    int row = tid >> 3, seg = tid & 7;
#pragma unroll
    for (int k = 0; k < 8; ++k)
      async_copy16(&Qs[row * LDQ + seg * 64 + k * 8],
                   src + (size_t)row * Cn + seg * 64 + k * 8);
  }
  if (tid < 32) { row_m[tid] = -1e30f; row_l[tid] = 0.f; }
  async_wait0();
  __syncthreads();

  v8f o[8];
#pragma unroll
  for (int t = 0; t < 8; ++t)
#pragma unroll
    for (int r = 0; r < 8; ++r) o[t][r] = 0.f;

  for (int j0 = 0; j0 < Ln; j0 += 64) {
    // prefetch next K/V j-tile into cache while we compute this one
    if (j0 + 64 < Ln) {
      const __bf16* knext = Kg + ((size_t)(b * Ln + j0 + 64 + ni * 16)) * Cn;
      const __bf16* vnext = Vt + ((size_t)(b * Cn + ni * 128)) * Ln + j0 + 64;
#pragma unroll
      for (int p = 0; p < 4; ++p) {
        int u = lane + p * 32;                         // 128 cache lines each
        __builtin_prefetch(knext + (size_t)(u >> 3) * Cn + (u & 7) * 64, 0, 1);
        __builtin_prefetch(vnext + (size_t)u * Ln, 0, 1);
      }
    }

    // ---- S tile (16x16 per wave), K = 512 ----
    v8f s;
#pragma unroll
    for (int r = 0; r < 8; ++r) s[r] = 0.f;
    const __bf16* qb = &Qs[mi * 16 * LDQ];
    const __bf16* kb = Kg + ((size_t)(b * Ln + j0 + ni * 16)) * Cn;
#pragma unroll 4
    for (int kk = 0; kk < Cn; kk += 32)
      s = wmma_bf16(load_a_frag(qb + kk, LDQ), load_bt_frag(kb + kk, Cn), s);

    // per-row max within this 16-wide tile (16-lane half reduction)
#pragma unroll
    for (int r = 0; r < 8; ++r) {
      float v = s[r];
      v = fmaxf(v, __shfl_xor(v, 1, 32));
      v = fmaxf(v, __shfl_xor(v, 2, 32));
      v = fmaxf(v, __shfl_xor(v, 4, 32));
      v = fmaxf(v, __shfl_xor(v, 8, 32));
      if ((lane & 15) == 0) red_m[mi * 16 + r + half * 8][ni] = v;
    }
    __syncthreads();

    if (w < 2 && lane < 16) {          // combine maxes, compute alpha
      int row = w * 16 + lane;
      float mp = row_m[row];
      float mn = fmaxf(fmaxf(red_m[row][0], red_m[row][1]),
                       fmaxf(red_m[row][2], red_m[row][3]));
      mn = fmaxf(mn, mp);
      row_a[row] = __expf(mp - mn);
      row_m[row] = mn;
    }
    __syncthreads();

    float mloc[8], aloc[8];
#pragma unroll
    for (int r = 0; r < 8; ++r) {
      int row = mi * 16 + r + half * 8;
      mloc[r] = row_m[row];
      aloc[r] = row_a[row];
    }
#pragma unroll
    for (int r = 0; r < 8; ++r) {      // exp, P -> LDS, partial row sums
      float e = __expf(s[r] - mloc[r]);
      Ps[(mi * 16 + r + half * 8) * LDP + ni * 16 + (lane & 15)] = (__bf16)e;
      float t = e;
      t += __shfl_xor(t, 1, 32);
      t += __shfl_xor(t, 2, 32);
      t += __shfl_xor(t, 4, 32);
      t += __shfl_xor(t, 8, 32);
      if ((lane & 15) == 0) red_s[mi * 16 + r + half * 8][ni] = t;
    }
#pragma unroll
    for (int t = 0; t < 8; ++t)        // rescale running O by alpha
#pragma unroll
      for (int r = 0; r < 8; ++r) o[t][r] *= aloc[r];
    __syncthreads();

    if (w < 2 && lane < 16) {          // update running denominator
      int row = w * 16 + lane;
      row_l[row] = row_l[row] * row_a[row] +
                   red_s[row][0] + red_s[row][1] + red_s[row][2] + red_s[row][3];
    }

    // ---- O += P(16x64) * V(64x128 slab) ----
    const __bf16* pb = &Ps[mi * 16 * LDP];
    bfrag pa0 = load_a_frag(pb, LDP);
    bfrag pa1 = load_a_frag(pb + 32, LDP);
#pragma unroll
    for (int t = 0; t < 8; ++t) {
      const __bf16* vb = Vt + ((size_t)(b * Cn + ni * 128 + t * 16)) * Ln + j0;
      o[t] = wmma_bf16(pa0, load_bt_frag(vb, Ln), o[t]);
      o[t] = wmma_bf16(pa1, load_bt_frag(vb + 32, Ln), o[t]);
    }
    __syncthreads();
  }

  // ---- epilogue: O / l -> bf16 [B,L,C] ----
  float linv[8];
#pragma unroll
  for (int r = 0; r < 8; ++r)
    linv[r] = 1.f / row_l[mi * 16 + r + half * 8];
#pragma unroll
  for (int t = 0; t < 8; ++t) {
    int col = ni * 128 + t * 16 + (lane & 15);
#pragma unroll
    for (int r = 0; r < 8; ++r) {
      int row = i0 + mi * 16 + r + half * 8;
      Og[((size_t)(b * Ln) + row) * Cn + col] = (__bf16)(o[t][r] * linv[r]);
    }
  }
}

extern "C" void kernel_launch(void* const* d_in, const int* in_sizes, int n_in,
                              void* d_out, int out_size, void* d_ws, size_t ws_size,
                              hipStream_t stream) {
  const float* x   = (const float*)d_in[0];
  const float* gnw = (const float*)d_in[1];
  const float* gnb = (const float*)d_in[2];
  const float* qw  = (const float*)d_in[3];
  const float* qb  = (const float*)d_in[4];
  const float* kw  = (const float*)d_in[5];
  const float* kb  = (const float*)d_in[6];
  const float* vw  = (const float*)d_in[7];
  const float* vb  = (const float*)d_in[8];
  const float* pw  = (const float*)d_in[9];
  const float* pb  = (const float*)d_in[10];
  float* out = (float*)d_out;

  char* ws = (char*)d_ws;
  const size_t MB = 1024 * 1024;
  __bf16* xnT = (__bf16*)(ws);            // [B,L,C]; reused as attention O
  __bf16* Qb_ = (__bf16*)(ws + 16 * MB);  // [B,L,C]
  __bf16* Kb_ = (__bf16*)(ws + 32 * MB);  // [B,L,C]
  __bf16* Vb_ = (__bf16*)(ws + 48 * MB);  // [B,C,L]
  __bf16* wqb = (__bf16*)(ws + 64 * MB);
  __bf16* wkb = wqb + 512 * 512;
  __bf16* wvb = wkb + 512 * 512;
  __bf16* wpb = wvb + 512 * 512;

  cvt_kernel<<<1024, 256, 0, stream>>>(qw, wqb, 512 * 512);
  cvt_kernel<<<1024, 256, 0, stream>>>(kw, wkb, 512 * 512);
  cvt_kernel<<<1024, 256, 0, stream>>>(vw, wvb, 512 * 512);
  cvt_kernel<<<1024, 256, 0, stream>>>(pw, wpb, 512 * 512);

  gn_kernel<<<Bn * Gn, 256, 0, stream>>>(x, gnw, gnb, xnT);

  const float att_scale = 0.044194173824159216f;  // 512^-0.5, folded into Q
  gemm_kernel<<<dim3(32, 8, Bn), 256, 0, stream>>>(xnT, wqb, qb, Qb_, nullptr,
                                                   nullptr, att_scale, 0);
  gemm_kernel<<<dim3(32, 8, Bn), 256, 0, stream>>>(xnT, wkb, kb, Kb_, nullptr,
                                                   nullptr, 1.f, 1);
  gemm_kernel<<<dim3(32, 8, Bn), 256, 0, stream>>>(xnT, wvb, vb, Vb_, nullptr,
                                                   nullptr, 1.f, 2);

  flash_kernel<<<dim3(Ln / 32, Bn), 256, 0, stream>>>(Qb_, Kb_, Vb_, xnT);

  gemm_kernel<<<dim3(32, 8, Bn), 256, 0, stream>>>(xnT, wpb, pb, nullptr, out,
                                                   x, 1.f, 3);
  (void)in_sizes; (void)n_in; (void)out_size; (void)ws_size;
}
